// ASAAttention_29308856827987
// MI455X (gfx1250) — compile-verified, matrix operand order checked by reading, not run on hardware
//
#include <hip/hip_runtime.h>
#include <hip/hip_bf16.h>
#include <math.h>
#include <stdint.h>

// ---------------------------------------------------------------------------
// ASA attention for MI455X (gfx1250, wave32, WMMA f16 -> f32 accumulate)
//   B=2, S=2048, D=1024, H=16, DK=64, FD=64
//   GEMM tiles staged by the Tensor Data Mover (tensor_load_to_lds),
//   flash staging via global_load_async_to_lds_b128.
// ---------------------------------------------------------------------------

#define B_  2
#define S_  2048
#define D_  1024
#define H_  16
#define DK_ 64
#define FD_ 64

typedef _Float16 half_t;
typedef __attribute__((ext_vector_type(16))) _Float16 v16h;
typedef __attribute__((ext_vector_type(8)))  _Float16 v8h;
typedef __attribute__((ext_vector_type(8)))  float    v8f;
typedef __attribute__((ext_vector_type(4)))  uint32_t u32x4;
typedef __attribute__((ext_vector_type(8)))  uint32_t u32x8;

static __device__ __forceinline__ v16h mk16(v8h lo, v8h hi) {
    v16h r;
#pragma unroll
    for (int i = 0; i < 8; ++i) { r[i] = lo[i]; r[i + 8] = hi[i]; }
    return r;
}

static __device__ __forceinline__ v8f zero8() {
    v8f z;
#pragma unroll
    for (int i = 0; i < 8; ++i) z[i] = 0.0f;
    return z;
}

// ---------------------------------------------------------------------------
// TDM: issue a 2-D tile load (tile0 x tile1 elements of 2 bytes) into LDS.
// Descriptor built in SGPRs (all arguments are wave-uniform).
//   group0: count=1 | lds_addr | global_addr(57b) | type=2
//   group1: data_size=2B | tensor dims | tile dims | dim0 stride
// ---------------------------------------------------------------------------
static __device__ __forceinline__ void tdm_load_2d(uint32_t lds_off,
                                                   const half_t* gptr,
                                                   uint32_t tdim0, uint32_t tdim1,
                                                   uint32_t tile0, uint32_t tile1,
                                                   uint32_t stride0) {
    uint64_t ga = (uint64_t)(uintptr_t)gptr;
    u32x4 d0;
    d0[0] = 1u;                                              // count=1 (valid)
    d0[1] = lds_off;                                         // lds_addr
    d0[2] = (uint32_t)ga;                                    // global_addr lo
    d0[3] = (uint32_t)((ga >> 32) & 0x01FFFFFFu) | (2u << 30); // ga hi | type=2
    u32x8 d1;
    d1[0] = (1u << 16);                                      // data_size = 2B
    d1[1] = (tdim0 & 0xFFFFu) << 16;                         // tensor_dim0 lo16
    d1[2] = (tdim0 >> 16) | ((tdim1 & 0xFFFFu) << 16);       // dim0 hi | dim1 lo
    d1[3] = (tdim1 >> 16) | (tile0 << 16);                   // dim1 hi | tile_dim0
    d1[4] = tile1;                                           // tile_dim1 (tile_dim2=0)
    d1[5] = stride0;                                         // tensor_dim0_stride lo32
    d1[6] = 0u;                                              // stride0 hi | stride1 lo
    d1[7] = 0u;
    asm volatile("tensor_load_to_lds %0, %1" :: "s"(d0), "s"(d1) : "memory");
}

// Per-lane async 16-byte copy global -> LDS (ASYNCcnt tracked).
static __device__ __forceinline__ void async_copy16(uint32_t lds_off,
                                                    const half_t* g) {
    asm volatile("global_load_async_to_lds_b128 %0, %1, off"
                 :: "v"(lds_off), "v"((uint64_t)(uintptr_t)g) : "memory");
}

// ---------------------------------------------------------------------------
// fp32 -> fp16 conversion (grid-stride)
// ---------------------------------------------------------------------------
__global__ void cvt_f16_kernel(const float* __restrict__ src,
                               half_t* __restrict__ dst, int n) {
    int i = blockIdx.x * blockDim.x + threadIdx.x;
    int stride = gridDim.x * blockDim.x;
    for (; i < n; i += stride) dst[i] = (half_t)src[i];
}

// ---------------------------------------------------------------------------
// Tiled f32->f16 transpose: dst[c][r] = src[r][c]
// ---------------------------------------------------------------------------
__global__ __launch_bounds__(256)
void transpose_cvt_kernel(const float* __restrict__ src,
                          half_t* __restrict__ dst, int rows, int cols) {
    __shared__ half_t t[32][33];
    int tx = threadIdx.x & 31;
    int ty = (threadIdx.x >> 5) << 2;      // 0,4,...,28
    int rb = blockIdx.y * 32;
    int cb = blockIdx.x * 32;
#pragma unroll
    for (int i = 0; i < 4; ++i)
        t[ty + i][tx] = (half_t)src[(size_t)(rb + ty + i) * cols + cb + tx];
    __syncthreads();
#pragma unroll
    for (int i = 0; i < 4; ++i)
        dst[(size_t)(cb + ty + i) * rows + rb + tx] = t[tx][ty + i];
}

// ---------------------------------------------------------------------------
// row-wise L2 normalize (rows of 64 floats) -> f16
// ---------------------------------------------------------------------------
__global__ void l2norm_f16_kernel(const float* __restrict__ src,
                                  half_t* __restrict__ dst, int rows) {
    int r = blockIdx.x * blockDim.x + threadIdx.x;
    if (r >= rows) return;
    const float* p = src + (size_t)r * FD_;
    float ss = 0.0f;
#pragma unroll
    for (int i = 0; i < FD_; ++i) ss += p[i] * p[i];
    float inv = 1.0f / fmaxf(sqrtf(ss), 1e-12f);
    half_t* q = dst + (size_t)r * FD_;
#pragma unroll
    for (int i = 0; i < FD_; ++i) q[i] = (half_t)(p[i] * inv);
}

// ---------------------------------------------------------------------------
// Tiled WMMA GEMM:  C[M,N] = (A[M,K] @ Bt[N,K]^T + bias[N]) * scale
//   A, Bt in f16 row-major (Bt pre-transposed so both tiles are contiguous).
//   Block tile 128x128, 8 waves (2x4), wave tile 64x32, K-step 32.
//   Tiles staged by TDM, double-buffered: DMA of tile k+1 overlaps compute.
// ---------------------------------------------------------------------------
template<int N, int K, bool OUT16>
__global__ __launch_bounds__(256)
void gemm_wmma_kernel(const half_t* __restrict__ A,
                      const half_t* __restrict__ Bt,
                      const float*  __restrict__ bias,
                      void* __restrict__ Cout,
                      int M, float scale) {
    __shared__ __align__(16) half_t sA[2][128 * 32];  // [m][k], 64B rows
    __shared__ __align__(16) half_t sB[2][128 * 32];  // [n][k], 64B rows

    const int tid  = threadIdx.x;
    const int lane = tid & 31;
    const int wid  = tid >> 5;
    const int wm   = wid >> 2;        // 0..1
    const int wn   = wid & 3;         // 0..3
    const int hsel = (lane >> 4) * 8; // K-half select per A/B VGPR striping
    const int lrow = lane & 15;
    const int hh   = lane >> 4;

    const int mbase = blockIdx.y * 128;
    const int nbase = blockIdx.x * 128;

    // Uniform (scalar-branch) wave-0 predicate for TDM issue.
    const bool isw0 = (__builtin_amdgcn_readfirstlane((int)threadIdx.x) >> 5) == 0;

    const half_t* gA = A  + (size_t)mbase * K;
    const half_t* gB = Bt + (size_t)nbase * K;
    const uint32_t ldsA0 = (uint32_t)(uintptr_t)&sA[0][0];
    const uint32_t ldsA1 = (uint32_t)(uintptr_t)&sA[1][0];
    const uint32_t ldsB0 = (uint32_t)(uintptr_t)&sB[0][0];
    const uint32_t ldsB1 = (uint32_t)(uintptr_t)&sB[1][0];

    if (isw0) {
        tdm_load_2d(ldsA0, gA, K, (uint32_t)M, 32, 128, K);
        tdm_load_2d(ldsB0, gB, K, N,          32, 128, K);
    }

    v8f acc[4][2];
#pragma unroll
    for (int mt = 0; mt < 4; ++mt)
#pragma unroll
        for (int nt = 0; nt < 2; ++nt) acc[mt][nt] = zero8();

    for (int k0 = 0; k0 < K; k0 += 32) {
        const int cur = (k0 >> 5) & 1;
        if (isw0) __builtin_amdgcn_s_wait_tensorcnt(0);
        __syncthreads();   // current tiles staged; previous tiles fully consumed
        if ((k0 + 32 < K) && isw0) {
            tdm_load_2d(cur ? ldsA0 : ldsA1, gA + k0 + 32, K, (uint32_t)M, 32, 128, K);
            tdm_load_2d(cur ? ldsB0 : ldsB1, gB + k0 + 32, K, N,          32, 128, K);
        }

        v16h af[4];
#pragma unroll
        for (int mt = 0; mt < 4; ++mt) {
            int row = wm * 64 + mt * 16 + lrow;
            v8h lo = *(const v8h*)&sA[cur][row * 32 + hsel];
            v8h hi = *(const v8h*)&sA[cur][row * 32 + hsel + 16];
            af[mt] = mk16(lo, hi);
        }
        v16h bf[2];
#pragma unroll
        for (int nt = 0; nt < 2; ++nt) {
            int col = wn * 32 + nt * 16 + lrow;
            v8h lo = *(const v8h*)&sB[cur][col * 32 + hsel];
            v8h hi = *(const v8h*)&sB[cur][col * 32 + hsel + 16];
            bf[nt] = mk16(lo, hi);
        }
#pragma unroll
        for (int mt = 0; mt < 4; ++mt)
#pragma unroll
            for (int nt = 0; nt < 2; ++nt)
                acc[mt][nt] = __builtin_amdgcn_wmma_f32_16x16x32_f16(
                    false, af[mt], false, bf[nt], (short)0, acc[mt][nt],
                    false, false);
    }

    // Epilogue: static N -> base address + immediate offsets.
#pragma unroll
    for (int nt = 0; nt < 2; ++nt) {
        const int gn = nbase + wn * 32 + nt * 16 + lrow;
        const float bv = bias[gn];
        const size_t base = (size_t)(mbase + wm * 64 + 8 * hh) * N + gn;
#pragma unroll
        for (int mt = 0; mt < 4; ++mt) {
#pragma unroll
            for (int v = 0; v < 8; ++v) {
                float val = (acc[mt][nt][v] + bv) * scale;
                if constexpr (OUT16)
                    ((half_t*)Cout)[base + (size_t)(mt * 16 + v) * N] = (half_t)val;
                else
                    ((float*)Cout)[base + (size_t)(mt * 16 + v) * N] = val;
            }
        }
    }
}

// ---------------------------------------------------------------------------
// Flash attention with ASA bias.
//   Q pre-scaled by 1/sqrt(DK). scores = Q.K^T + rn.fn^T, causal,
//   online softmax, out = P.V -> attnh (f16, [B,S,H*DK]).
//   Grid: (S/128, H, B); 256 threads = 8 waves; wave owns 16 query rows.
//   K / feature chunks staged via global_load_async_to_lds_b128.
// ---------------------------------------------------------------------------
__global__ __launch_bounds__(256)
void flash_asa_kernel(const half_t* __restrict__ Qh,
                      const half_t* __restrict__ Kh,
                      const half_t* __restrict__ Vh,
                      const half_t* __restrict__ rnh,
                      const half_t* __restrict__ fnh,
                      half_t* __restrict__ attnh) {
    __shared__ __align__(16) half_t sK[32 * 64];     // [key][d]   4KB
    __shared__ __align__(16) half_t sF[32 * 64];     // [key][f]   4KB
    __shared__ __align__(16) half_t sVt[64 * 32];    // [d][key]   4KB
    __shared__ __align__(16) half_t sP[8 * 16 * 32]; // per-wave P 8KB

    const int tid  = threadIdx.x;
    const int lane = tid & 31;
    const int wid  = tid >> 5;
    const int hsel = (lane >> 4) * 8;
    const int lrow = lane & 15;
    const int hh   = lane >> 4;

    const int h     = blockIdx.y;
    const int b     = blockIdx.z;
    const int qbase = blockIdx.x * 128;
    const int qrow0 = qbase + wid * 16;

    const float NEG = -1.0e30f;

    // Loop-invariant A fragments: 16 query rows x 64 (DK / FD)
    v16h qf[2], rf[2];
    {
        size_t qoff = ((size_t)(b * S_ + qrow0 + lrow)) * D_ + h * DK_;
#pragma unroll
        for (int kc = 0; kc < 2; ++kc) {
            v8h lo = *(const v8h*)(Qh + qoff + kc * 32 + hsel);
            v8h hi = *(const v8h*)(Qh + qoff + kc * 32 + hsel + 16);
            qf[kc] = mk16(lo, hi);
        }
        size_t roff = ((size_t)(b * S_ + qrow0 + lrow)) * FD_;
#pragma unroll
        for (int kc = 0; kc < 2; ++kc) {
            v8h lo = *(const v8h*)(rnh + roff + kc * 32 + hsel);
            v8h hi = *(const v8h*)(rnh + roff + kc * 32 + hsel + 16);
            rf[kc] = mk16(lo, hi);
        }
    }

    v8f oacc[4];
#pragma unroll
    for (int nt = 0; nt < 4; ++nt) oacc[nt] = zero8();
    float m[8], l[8];
#pragma unroll
    for (int v = 0; v < 8; ++v) { m[v] = NEG; l[v] = 0.0f; }

    const int nchunks = (qbase + 128) / 32;
    for (int c = 0; c < nchunks; ++c) {
        const int t0 = c * 32;
        __syncthreads();
        // Cooperative staging: async K + feature copies, manual V transpose.
        {
            int key = tid >> 3;
            int dc  = (tid & 7) * 8;
            size_t krow = (size_t)(b * S_ + t0 + key);
            async_copy16((uint32_t)(uintptr_t)&sK[key * 64 + dc],
                         Kh + krow * D_ + h * DK_ + dc);
            async_copy16((uint32_t)(uintptr_t)&sF[key * 64 + dc],
                         fnh + krow * FD_ + dc);
            v8h vv = *(const v8h*)(Vh + krow * D_ + h * DK_ + dc);
#pragma unroll
            for (int e = 0; e < 8; ++e) sVt[(dc + e) * 32 + key] = vv[e];
        }
        asm volatile("s_wait_asynccnt 0x0" ::: "memory");
        __syncthreads();

        // Scores for two 16-key tiles: bias WMMAs chained into QK WMMAs.
        v8f sc[2];
#pragma unroll
        for (int tile = 0; tile < 2; ++tile) {
            int key = tile * 16 + lrow;
            v8f a = zero8();
#pragma unroll
            for (int kc = 0; kc < 2; ++kc) {
                v8h lo = *(const v8h*)&sF[key * 64 + kc * 32 + hsel];
                v8h hi = *(const v8h*)&sF[key * 64 + kc * 32 + hsel + 16];
                a = __builtin_amdgcn_wmma_f32_16x16x32_f16(
                        false, rf[kc], false, mk16(lo, hi), (short)0, a,
                        false, false);
            }
#pragma unroll
            for (int kc = 0; kc < 2; ++kc) {
                v8h lo = *(const v8h*)&sK[key * 64 + kc * 32 + hsel];
                v8h hi = *(const v8h*)&sK[key * 64 + kc * 32 + hsel + 16];
                a = __builtin_amdgcn_wmma_f32_16x16x32_f16(
                        false, qf[kc], false, mk16(lo, hi), (short)0, a,
                        false, false);
            }
            int kg = t0 + tile * 16 + lrow;
#pragma unroll
            for (int v = 0; v < 8; ++v) {
                int qg = qrow0 + v + 8 * hh;
                if (kg > qg) a[v] = NEG;   // causal mask
            }
            sc[tile] = a;
        }

        // Online softmax per row; row = v + 8*hh matches the C-layout.
#pragma unroll
        for (int v = 0; v < 8; ++v) {
            float s0 = sc[0][v], s1 = sc[1][v];
            float mx = fmaxf(s0, s1);
#pragma unroll
            for (int o = 8; o >= 1; o >>= 1)
                mx = fmaxf(mx, __shfl_xor(mx, o, 32));
            float mnew = fmaxf(m[v], mx);
            float corr = __expf(m[v] - mnew);
            float p0 = __expf(s0 - mnew);
            float p1 = __expf(s1 - mnew);
            float rs = p0 + p1;
#pragma unroll
            for (int o = 8; o >= 1; o >>= 1)
                rs += __shfl_xor(rs, o, 32);
            l[v] = l[v] * corr + rs;
            m[v] = mnew;
#pragma unroll
            for (int nt = 0; nt < 4; ++nt) oacc[nt][v] *= corr;
            int prow = v + 8 * hh;
            sP[wid * 512 + prow * 32 + lrow]      = (half_t)p0;
            sP[wid * 512 + prow * 32 + 16 + lrow] = (half_t)p1;
        }
        asm volatile("s_wait_dscnt 0x0" ::: "memory");  // wave-local LDS RAW

        // P (16x32 f16 A-frag) @ V chunk (32x64)
        v8h plo = *(const v8h*)&sP[wid * 512 + lrow * 32 + hsel];
        v8h phi = *(const v8h*)&sP[wid * 512 + lrow * 32 + hsel + 16];
        v16h pf = mk16(plo, phi);
#pragma unroll
        for (int nt = 0; nt < 4; ++nt) {
            int dk = nt * 16 + lrow;
            v8h lo = *(const v8h*)&sVt[dk * 32 + hsel];
            v8h hi = *(const v8h*)&sVt[dk * 32 + hsel + 16];
            oacc[nt] = __builtin_amdgcn_wmma_f32_16x16x32_f16(
                    false, pf, false, mk16(lo, hi), (short)0, oacc[nt],
                    false, false);
        }
    }

    // Epilogue: normalize and write f16 attention output.
#pragma unroll
    for (int nt = 0; nt < 4; ++nt) {
#pragma unroll
        for (int v = 0; v < 8; ++v) {
            int qg = qrow0 + v + 8 * hh;
            float val = oacc[nt][v] / l[v];
            attnh[((size_t)(b * S_ + qg)) * D_ + h * DK_ + nt * 16 + lrow] =
                (half_t)val;
        }
    }
}

// ---------------------------------------------------------------------------
// Host-side orchestration
// ---------------------------------------------------------------------------
extern "C" void kernel_launch(void* const* d_in, const int* in_sizes, int n_in,
                              void* d_out, int out_size, void* d_ws, size_t ws_size,
                              hipStream_t stream) {
    (void)in_sizes; (void)n_in; (void)out_size; (void)ws_size;

    const float* x    = (const float*)d_in[0];
    const float* feat = (const float*)d_in[1];
    const float* req  = (const float*)d_in[2];
    const float* Wq   = (const float*)d_in[3];
    const float* bq   = (const float*)d_in[4];
    const float* Wk   = (const float*)d_in[5];
    const float* bk   = (const float*)d_in[6];
    const float* Wv   = (const float*)d_in[7];
    const float* bv   = (const float*)d_in[8];
    const float* Wo   = (const float*)d_in[9];
    const float* bo   = (const float*)d_in[10];
    // d_in[11] = pos_ids, d_in[12] = causal_mask: unused (causality explicit)

    char* w = (char*)d_ws;
    half_t* xh  = (half_t*)(w + ((size_t)0  << 20));  // 8MB
    half_t* wqh = (half_t*)(w + ((size_t)8  << 20));  // 2MB (transposed)
    half_t* wkh = (half_t*)(w + ((size_t)10 << 20));  // 2MB (transposed)
    half_t* wvh = (half_t*)(w + ((size_t)12 << 20));  // 2MB (transposed)
    half_t* woh = (half_t*)(w + ((size_t)14 << 20));  // 2MB (transposed)
    half_t* qh  = (half_t*)(w + ((size_t)16 << 20));  // 8MB
    half_t* kh  = (half_t*)(w + ((size_t)24 << 20));  // 8MB
    half_t* vh  = (half_t*)(w + ((size_t)32 << 20));  // 8MB
    half_t* ah  = (half_t*)(w + ((size_t)40 << 20));  // 8MB
    half_t* rnh = (half_t*)(w + ((size_t)48 << 20));  // 0.5MB
    half_t* fnh = (half_t*)(w + ((size_t)49 << 20));  // 0.5MB

    const int MN = B_ * S_;      // 4096
    const float qscale = 0.125f; // 1/sqrt(DK)

    cvt_f16_kernel<<<1024, 256, 0, stream>>>(x, xh, MN * D_);
    dim3 tgrid(D_ / 32, D_ / 32);
    transpose_cvt_kernel<<<tgrid, 256, 0, stream>>>(Wq, wqh, D_, D_);
    transpose_cvt_kernel<<<tgrid, 256, 0, stream>>>(Wk, wkh, D_, D_);
    transpose_cvt_kernel<<<tgrid, 256, 0, stream>>>(Wv, wvh, D_, D_);
    transpose_cvt_kernel<<<tgrid, 256, 0, stream>>>(Wo, woh, D_, D_);
    l2norm_f16_kernel<<<(MN + 255) / 256, 256, 0, stream>>>(req,  rnh, MN);
    l2norm_f16_kernel<<<(MN + 255) / 256, 256, 0, stream>>>(feat, fnh, MN);

    dim3 ggrid(D_ / 128, MN / 128);  // (8, 32)
    gemm_wmma_kernel<D_, D_, true><<<ggrid, 256, 0, stream>>>(
        xh, wqh, bq, qh, MN, qscale);
    gemm_wmma_kernel<D_, D_, true><<<ggrid, 256, 0, stream>>>(
        xh, wkh, bk, kh, MN, 1.0f);
    gemm_wmma_kernel<D_, D_, true><<<ggrid, 256, 0, stream>>>(
        xh, wvh, bv, vh, MN, 1.0f);

    dim3 fgrid(S_ / 128, H_, B_);    // (16, 16, 2)
    flash_asa_kernel<<<fgrid, 256, 0, stream>>>(qh, kh, vh, rnh, fnh, ah);

    gemm_wmma_kernel<D_, D_, false><<<ggrid, 256, 0, stream>>>(
        ah, woh, bo, (float*)d_out, MN, 1.0f);
}
